// SwitchMoE_78924319031887
// MI455X (gfx1250) — matrix operand order repaired and checked by reading, not compiled, not run on hardware
//
#include <hip/hip_runtime.h>
#include <hip/hip_bf16.h>
#include <stdint.h>

#define NTOK 16384
#define DDIM 512
#define FDIM 2048
#define NEXP 8

typedef __attribute__((ext_vector_type(16))) __bf16 bfx16;
typedef __attribute__((ext_vector_type(8)))  float  fx8;

union Frag { uint4 u[2]; bfx16 v; };

__device__ __forceinline__ unsigned short f2bf(float f) {
    union { float f; unsigned u; } v; v.f = f;
    unsigned r = v.u + 0x7FFFu + ((v.u >> 16) & 1u);   // round-to-nearest-even
    return (unsigned short)(r >> 16);
}

__device__ __forceinline__ fx8 zero8() {
    fx8 z;
#pragma unroll
    for (int k = 0; k < 8; ++k) z[k] = 0.0f;
    return z;
}

// ---------------- Router: top-1 over logits (== argmax of softmax) ----------------
__global__ void k_router(const float* __restrict__ x, const float* __restrict__ Wg,
                         int* __restrict__ top1, float* __restrict__ topval,
                         int* __restrict__ counts) {
    int n = blockIdx.x * blockDim.x + threadIdx.x;
    if (n >= NTOK) return;
    const float4* x4 = (const float4*)(x + (size_t)n * DDIM);
    const float4* w4 = (const float4*)Wg;
    float acc[NEXP];
#pragma unroll
    for (int e = 0; e < NEXP; ++e) acc[e] = 0.0f;
    for (int d = 0; d < DDIM / 4; ++d) {
        float4 xv = x4[d];
#pragma unroll
        for (int e = 0; e < NEXP; ++e) {
            float4 wv = w4[e * (DDIM / 4) + d];
            acc[e] += xv.x * wv.x + xv.y * wv.y + xv.z * wv.z + xv.w * wv.w;
        }
    }
    int be = 0; float bv = acc[0];
#pragma unroll
    for (int e = 1; e < NEXP; ++e) if (acc[e] > bv) { bv = acc[e]; be = e; }
    top1[n] = be;
    topval[n] = bv;                 // masked = gate_logits * one_hot -> the raw logit
    atomicAdd(&counts[be], 1);      // integer atomic: deterministic
}

// ---------------- Deterministic per-expert denominator reduction ----------------
__global__ void k_denom(const int* __restrict__ top1, const float* __restrict__ topval,
                        float* __restrict__ denom) {
    __shared__ float red[256];
    float acc[NEXP];
#pragma unroll
    for (int e = 0; e < NEXP; ++e) acc[e] = 0.0f;
    for (int n = threadIdx.x; n < NTOK; n += 256)
        acc[top1[n]] += topval[n];
    for (int e = 0; e < NEXP; ++e) {
        red[threadIdx.x] = acc[e];
        __syncthreads();
        for (int s = 128; s > 0; s >>= 1) {
            if ((int)threadIdx.x < s) red[threadIdx.x] += red[threadIdx.x + s];
            __syncthreads();
        }
        if (threadIdx.x == 0) denom[e] = red[0];
        __syncthreads();
    }
}

// ---------------- Tiny exclusive scan over 8 experts ----------------
__global__ void k_scan(const int* __restrict__ counts, int* __restrict__ offs,
                       int* __restrict__ cursor) {
    if (threadIdx.x == 0 && blockIdx.x == 0) {
        int s = 0;
        for (int e = 0; e < NEXP; ++e) { offs[e] = s; cursor[e] = s; s += counts[e]; }
    }
}

// ---------------- Build permutation (token lists per expert) + gate scale ----------------
__global__ void k_build(const int* __restrict__ top1, const float* __restrict__ topval,
                        const float* __restrict__ denom, int* __restrict__ cursor,
                        int* __restrict__ perm, float* __restrict__ gscale) {
    int n = blockIdx.x * blockDim.x + threadIdx.x;
    if (n >= NTOK) return;
    int e = top1[n];
    int pos = atomicAdd(&cursor[e], 1);   // position varies run-to-run, output doesn't
    perm[pos] = n;
    gscale[n] = topval[n] / (denom[e] + 1e-6f) * (float)NTOK;   // capacity = N
}

// ---------------- fp32 -> bf16 conversion of x ----------------
__global__ void k_cvt_x(const float* __restrict__ x, unsigned short* __restrict__ xb) {
    int i = blockIdx.x * blockDim.x + threadIdx.x;
    if (i >= NTOK * DDIM / 8) return;
    const float4* s = (const float4*)x + (size_t)i * 2;
    float4 a = s[0], b = s[1];
    union { unsigned short h[8]; uint4 u; } o;
    o.h[0] = f2bf(a.x); o.h[1] = f2bf(a.y); o.h[2] = f2bf(a.z); o.h[3] = f2bf(a.w);
    o.h[4] = f2bf(b.x); o.h[5] = f2bf(b.y); o.h[6] = f2bf(b.z); o.h[7] = f2bf(b.w);
    ((uint4*)xb)[i] = o.u;
}

// ---------------- Pre-swizzle weights into WMMA B-fragment-major bf16 ----------------
// B fragment (32K x 16N, bf16): lane l -> Khalf = l>>4, Ncol = l&15; 16 contiguous K per lane.
// dst fragment index = (e*(Nc/16) + nt)*(K/32) + kt ; each fragment = 512 ushorts (1 KB).
__global__ void k_swizzleB(const float* __restrict__ src, unsigned short* __restrict__ dst,
                           int K, int Nc, int total) {
    int id = blockIdx.x * blockDim.x + threadIdx.x;
    if (id >= total) return;
    int lane = id & 31;
    int frag = id >> 5;
    int ktiles = K >> 5;
    int ntiles = Nc >> 4;
    int kt = frag % ktiles;
    int rest = frag / ktiles;
    int nt = rest % ntiles;
    int e  = rest / ntiles;
    int khalf = lane >> 4, ncol = lane & 15;
    union { unsigned short s[16]; uint4 u[2]; } buf;
#pragma unroll
    for (int j = 0; j < 16; ++j) {
        int d = kt * 32 + khalf * 16 + j;
        buf.s[j] = f2bf(src[((size_t)e * K + d) * (size_t)Nc + nt * 16 + ncol]);
    }
    uint4* dp = (uint4*)(dst + (size_t)frag * 512 + lane * 16);
    dp[0] = buf.u[0];
    dp[1] = buf.u[1];
}

// ---------------- Fused expert FFN: relu(x@W1+b1)@W2+b2, gated scatter ----------------
// Block: 256 threads = 8 waves. Tile: 64 tokens x full D=512 output for one expert.
// Wave grid: mi = w&3 (16-row slice), ng = w>>2 (256-col half of D).
// Padding rows map to the all-zero row xb[NTOK] -> no divergence in the k-loop.
__global__ __launch_bounds__(256) void k_ffn(
    const unsigned short* __restrict__ xb,
    const unsigned short* __restrict__ w1b,
    const unsigned short* __restrict__ w2b,
    const float* __restrict__ b1, const float* __restrict__ b2,
    const int* __restrict__ perm, const int* __restrict__ offs,
    const int* __restrict__ counts, const float* __restrict__ gscale,
    float* __restrict__ out)
{
    __shared__ unsigned short hA[64 * 128];   // bf16 h-chunk, 16 KB
    const int e = blockIdx.y;
    const int t = blockIdx.x;
    const int cnt = counts[e];
    if (t * 64 >= cnt) return;                // uniform exit; EXEC stays all-ones below
    const int segbase = offs[e];
    const int tid = threadIdx.x;
    const int w = tid >> 5, l = tid & 31;
    const int mi = w & 3, ng = w >> 2;
    const int lm = l & 15, lh = l >> 4;

    // Token for this lane's A-matrix row (M = lane%16, same for both half-waves).
    // Invalid rows -> zero row at index NTOK (unconditional loads, full EXEC).
    const int rA = t * 64 + mi * 16 + lm;
    const int tokA = (rA < cnt) ? perm[segbase + rA] : NTOK;
    const unsigned short* xrow = xb + (size_t)tokA * DDIM;

    fx8 acc2[16];
#pragma unroll
    for (int i = 0; i < 16; ++i) acc2[i] = zero8();

#pragma unroll 1
    for (int fc = 0; fc < FDIM; fc += 128) {
        // ---- GEMM1: hc[64x128] = relu(x_tile @ W1[:, fc:fc+128] + b1) ----
        fx8 acc1[4];
#pragma unroll
        for (int j = 0; j < 4; ++j) acc1[j] = zero8();

#pragma unroll 2
        for (int kk = 0; kk < DDIM; kk += 32) {
            const uint4* p = (const uint4*)(xrow + kk + lh * 16);
            Frag A;
            A.u[0] = p[0];
            A.u[1] = p[1];                // +8 bf16 elements = +16 B
#pragma unroll
            for (int j = 0; j < 4; ++j) {
                int ntg = (fc >> 4) + ng * 4 + j;
                const uint4* bp = (const uint4*)(w1b +
                    ((size_t)(e * (FDIM / 16) + ntg) * (DDIM / 32) + (kk >> 5)) * 512 + l * 16);
                Frag B; B.u[0] = bp[0]; B.u[1] = bp[1];
                acc1[j] = __builtin_amdgcn_wmma_f32_16x16x32_bf16(
                    false, A.v, false, B.v, (short)0, acc1[j], false, false);
            }
        }
        // bias + ReLU + bf16 + LDS store (row-major [64][128])
#pragma unroll
        for (int j = 0; j < 4; ++j) {
            int lcol = (ng * 4 + j) * 16 + lm;
            float bb = b1[e * FDIM + fc + lcol];
#pragma unroll
            for (int r = 0; r < 8; ++r) {
                float v = acc1[j][r] + bb;
                v = v > 0.0f ? v : 0.0f;
                int m = mi * 16 + lh * 8 + r;
                hA[m * 128 + lcol] = f2bf(v);
            }
        }
        __syncthreads();

        // ---- GEMM2: acc2[64x512] += hc @ W2[fc:fc+128, :] ----
#pragma unroll 1
        for (int kt = 0; kt < 4; ++kt) {
            const uint4* hp = (const uint4*)(&hA[(mi * 16 + lm) * 128 + kt * 32 + lh * 16]);
            Frag A;
            A.u[0] = hp[0];
            A.u[1] = hp[1];
#pragma unroll
            for (int i = 0; i < 16; ++i) {
                int ntD = ng * 16 + i;
                const uint4* bp = (const uint4*)(w2b +
                    ((size_t)(e * (DDIM / 16) + ntD) * (FDIM / 32) + (fc >> 5) + kt) * 512 + l * 16);
                Frag B; B.u[0] = bp[0]; B.u[1] = bp[1];
                acc2[i] = __builtin_amdgcn_wmma_f32_16x16x32_bf16(
                    false, A.v, false, B.v, (short)0, acc2[i], false, false);
            }
        }
        __syncthreads();   // before next chunk rewrites hA
    }

    // ---- Epilogue: + b2, * gate, scatter to out (each token written exactly once) ----
    int   toks[8];
    float gs[8];
    bool  val[8];
#pragma unroll
    for (int r = 0; r < 8; ++r) {
        int m = mi * 16 + lh * 8 + r;
        int rg = t * 64 + m;
        val[r] = rg < cnt;
        int tk = val[r] ? perm[segbase + rg] : 0;
        toks[r] = tk;
        gs[r] = val[r] ? gscale[tk] : 0.0f;
    }
#pragma unroll
    for (int i = 0; i < 16; ++i) {
        int dcol = (ng * 16 + i) * 16 + lm;
        float b2v = b2[e * DDIM + dcol];
#pragma unroll
        for (int r = 0; r < 8; ++r) {
            if (val[r])
                out[(size_t)toks[r] * DDIM + dcol] = (acc2[i][r] + b2v) * gs[r];
        }
    }
}

extern "C" void kernel_launch(void* const* d_in, const int* in_sizes, int n_in,
                              void* d_out, int out_size, void* d_ws, size_t ws_size,
                              hipStream_t stream) {
    const float* x  = (const float*)d_in[0];
    const float* Wg = (const float*)d_in[1];
    const float* W1 = (const float*)d_in[2];
    const float* b1 = (const float*)d_in[3];
    const float* W2 = (const float*)d_in[4];
    const float* b2 = (const float*)d_in[5];
    float* out = (float*)d_out;

    char* p = (char*)d_ws;
    auto alloc = [&](size_t bytes) -> char* {
        char* r = p; p += (bytes + 255) & ~(size_t)255; return r;
    };
    unsigned short* xb     = (unsigned short*)alloc((size_t)(NTOK + 1) * DDIM * 2); // +1 zero row
    unsigned short* w1b    = (unsigned short*)alloc((size_t)NEXP * DDIM * FDIM * 2);
    unsigned short* w2b    = (unsigned short*)alloc((size_t)NEXP * FDIM * DDIM * 2);
    int*            top1   = (int*)alloc((size_t)NTOK * 4);
    float*          topval = (float*)alloc((size_t)NTOK * 4);
    int*            perm   = (int*)alloc((size_t)NTOK * 4);
    float*          gscale = (float*)alloc((size_t)NTOK * 4);
    int*            counts = (int*)alloc(256);
    int*            offs   = (int*)alloc(256);
    int*            cursor = (int*)alloc(256);
    float*          denom  = (float*)alloc(256);
    (void)in_sizes; (void)n_in; (void)out_size; (void)ws_size;

    hipMemsetAsync(counts, 0, 256, stream);
    hipMemsetAsync(xb + (size_t)NTOK * DDIM, 0, DDIM * 2, stream);   // zero pad row

    k_router<<<NTOK / 256, 256, 0, stream>>>(x, Wg, top1, topval, counts);
    k_denom <<<1, 256, 0, stream>>>(top1, topval, denom);
    k_scan  <<<1, 32, 0, stream>>>(counts, offs, cursor);
    k_build <<<NTOK / 256, 256, 0, stream>>>(top1, topval, denom, cursor, perm, gscale);

    k_cvt_x <<<(NTOK * DDIM / 8 + 255) / 256, 256, 0, stream>>>(x, xb);
    {
        int totW1 = NEXP * (DDIM / 32) * (FDIM / 16) * 32;
        k_swizzleB<<<(totW1 + 255) / 256, 256, 0, stream>>>(W1, w1b, DDIM, FDIM, totW1);
        int totW2 = NEXP * (FDIM / 32) * (DDIM / 16) * 32;
        k_swizzleB<<<(totW2 + 255) / 256, 256, 0, stream>>>(W2, w2b, FDIM, DDIM, totW2);
    }

    dim3 grid(NTOK / 64, NEXP);
    k_ffn<<<grid, 256, 0, stream>>>(xb, w1b, w2b, b1, b2, perm, offs, counts, gscale, out);
}
